// MultiHeadAttention_85976655331536
// MI455X (gfx1250) — compile-verified, matrix operand order checked by reading, not run on hardware
//
#include <hip/hip_runtime.h>

typedef __attribute__((ext_vector_type(16))) __bf16 v16bf;
typedef __attribute__((ext_vector_type(8)))  float  v8f;

#define DEV __device__ __forceinline__

constexpr int Tq = 2048, DIN = 1024, DOUT = 1024, Hh = 16, HD = 64;

#if __has_builtin(__builtin_amdgcn_global_load_async_to_lds_b128) && \
    __has_builtin(__builtin_amdgcn_s_wait_asynccnt)
#define HAS_ASYNC_LDS 1
typedef int v4i __attribute__((vector_size(16)));
typedef v4i __attribute__((address_space(1))) gv4i;   // global
typedef v4i __attribute__((address_space(3))) lv4i;   // LDS
#else
#define HAS_ASYNC_LDS 0
#endif

union Frag16 { v16bf v; unsigned u[8]; unsigned short s[16]; };

DEV unsigned short f2bf(float x) {
  unsigned u = __builtin_bit_cast(unsigned, x);
  u += 0x7FFFu + ((u >> 16) & 1u);              // round-to-nearest-even
  return (unsigned short)(u >> 16);
}

// ISA 16-bit A/B fragment K mapping: VGPR v holds element pair starting at k.
DEV int frag_k(int v, int khalf) {
  return (v < 4) ? (khalf * 8 + v * 2) : (16 + khalf * 8 + (v - 4) * 2);
}

// Load a 16x32 A-fragment (or 32x16 B-fragment) of bf16; caller passes the
// per-lane row base pointer (row-major, k contiguous, 4B-aligned).
DEV v16bf load_frag(const unsigned short* rowbase) {
  Frag16 f;
  const int khalf = (threadIdx.x >> 4) & 1;
#pragma unroll
  for (int v = 0; v < 8; ++v) {
    const int k = frag_k(v, khalf);
    f.u[v] = *(const unsigned*)(rowbase + k);
  }
  return f.v;
}

DEV v8f wmma_bf16(v16bf a, v16bf b, v8f c) {
  return __builtin_amdgcn_wmma_f32_16x16x32_bf16(false, a, false, b,
                                                 (short)0, c, false, false);
}

// ---------------------------------------------------------------------------
// Stage 1: Y = X @ W + b, X f32 [8192,1024], W f32 [1024,1024],
// output bf16 scattered into [B, H, T, HD].
// Block: 256 thr (8 waves), tile 128x128, wave tile 32x64, K-step 32.
// ---------------------------------------------------------------------------
__global__ __launch_bounds__(256)
void proj_qkv_kernel(const float* __restrict__ X, const float* __restrict__ W,
                     const float* __restrict__ bias,
                     unsigned short* __restrict__ Out) {
  __shared__ unsigned short As[128][32];   // [m][k] bf16
  __shared__ unsigned short Bs[128][32];   // [n][k] bf16 (transposed W)

  const int tid = threadIdx.x;
  const int lane = tid & 31, wave = tid >> 5;
  const int lm = lane & 15, mhalf = lane >> 4;
  const int wm = (wave & 3) * 32;          // 4 waves tile 128 rows
  const int wn = (wave >> 2) * 64;         // 2 waves tile 128 cols
  const int m0 = blockIdx.y * 128, n0 = blockIdx.x * 128;

  const int arow = tid >> 1, acol = (tid & 1) * 16;
  const int wrow = tid >> 3, wcol = (tid & 7) * 16;

  v8f c[2][4] = {};

  for (int kk = 0; kk < DIN; kk += 32) {
    { // stage X tile (128x32) f32 -> bf16
      const float* src = X + (size_t)(m0 + arow) * DIN + kk + acol;
      if (kk + 32 < DIN) __builtin_prefetch(src + 32, 0, 0);
      float vals[16];
#pragma unroll
      for (int i = 0; i < 4; ++i) {
        float4 f = ((const float4*)src)[i];
        vals[i * 4 + 0] = f.x; vals[i * 4 + 1] = f.y;
        vals[i * 4 + 2] = f.z; vals[i * 4 + 3] = f.w;
      }
      unsigned* dst = (unsigned*)&As[arow][acol];
#pragma unroll
      for (int j = 0; j < 8; ++j)
        dst[j] = (unsigned)f2bf(vals[2 * j]) |
                 ((unsigned)f2bf(vals[2 * j + 1]) << 16);
    }
    { // stage W tile (32x128) f32 -> bf16, transposed to [n][k]
      const float* src = W + (size_t)(kk + wrow) * DOUT + n0 + wcol;
      float vals[16];
#pragma unroll
      for (int i = 0; i < 4; ++i) {
        float4 f = ((const float4*)src)[i];
        vals[i * 4 + 0] = f.x; vals[i * 4 + 1] = f.y;
        vals[i * 4 + 2] = f.z; vals[i * 4 + 3] = f.w;
      }
#pragma unroll
      for (int j = 0; j < 16; ++j) Bs[wcol + j][wrow] = f2bf(vals[j]);
    }
    __syncthreads();

    v16bf a0 = load_frag(&As[wm + lm][0]);
    v16bf a1 = load_frag(&As[wm + 16 + lm][0]);
#pragma unroll
    for (int j = 0; j < 4; ++j) {
      const v16bf b = load_frag(&Bs[wn + j * 16 + lm][0]);
      c[0][j] = wmma_bf16(a0, b, c[0][j]);
      c[1][j] = wmma_bf16(a1, b, c[1][j]);
    }
    __syncthreads();
  }

#pragma unroll
  for (int j = 0; j < 4; ++j) {
    const int gn = n0 + wn + j * 16 + lm;
    const float bj = bias[gn];
    const int h = gn >> 6, hd = gn & 63;
#pragma unroll
    for (int i = 0; i < 2; ++i) {
#pragma unroll
      for (int r = 0; r < 8; ++r) {
        const int gm = m0 + wm + i * 16 + r + 8 * mhalf;
        const int bb = gm >> 11, t = gm & (Tq - 1);
        Out[(size_t)((bb * Hh + h) * Tq + t) * HD + hd] =
            f2bf(c[i][j][r] + bj);
      }
    }
  }
}

// ---------------------------------------------------------------------------
// Stage 2: flash attention. Block = 4 waves = 64 query rows of one (b,h).
// Wave owns 16 q rows; key loop in 64-key blocks up to the causal diagonal.
// ---------------------------------------------------------------------------
__global__ __launch_bounds__(128)
void flash_attn_kernel(const unsigned short* __restrict__ Q,
                       const unsigned short* __restrict__ K,
                       const unsigned short* __restrict__ V,
                       unsigned short* __restrict__ Ctx) {
  __shared__ unsigned short Ks[64][64];      // [key][hd]
  __shared__ unsigned short Vt[64][64];      // [hd][key] (transposed)
  __shared__ unsigned short Ps[4][16][64];   // per-wave P transpose buffer

  const int tid = threadIdx.x;
  const int lane = tid & 31, wave = tid >> 5;
  const int lm = lane & 15, mhalf = lane >> 4;
  const int qb = blockIdx.x;                 // 64-row query block
  const int bh = blockIdx.y;
  const int bb = bh >> 4, h = bh & 15;

  const int qw0 = qb * 64 + wave * 16;       // first q row of this wave
  const unsigned short* Qbase = Q + ((size_t)bh * Tq + qw0 + lm) * HD;
  v16bf qa[2];
#pragma unroll
  for (int kc = 0; kc < 2; ++kc) qa[kc] = load_frag(Qbase + kc * 32);

  v8f o[4] = {};
  float row_max[8], row_sum[8];
#pragma unroll
  for (int r = 0; r < 8; ++r) { row_max[r] = -3.0e38f; row_sum[r] = 0.0f; }

  const int key = tid >> 1, half = (tid & 1) * 32;

  for (int kb = 0; kb <= qb; ++kb) {
    const int kg0 = kb * 64;
    { // stage K row-major (async DMA to LDS if available), V transposed (sync)
#if HAS_ASYNC_LDS
      gv4i* ksrc = (gv4i*)(K + ((size_t)bh * Tq + kg0 + key) * HD + half);
      lv4i* kdst = (lv4i*)&Ks[key][half];
      __builtin_amdgcn_global_load_async_to_lds_b128(ksrc, kdst, 0, 0);
      __builtin_amdgcn_global_load_async_to_lds_b128(ksrc, kdst, 16, 0);
      __builtin_amdgcn_global_load_async_to_lds_b128(ksrc, kdst, 32, 0);
      __builtin_amdgcn_global_load_async_to_lds_b128(ksrc, kdst, 48, 0);
#else
      const uint4* ksrc =
          (const uint4*)(K + ((size_t)bh * Tq + kg0 + key) * HD + half);
      uint4* kdst = (uint4*)&Ks[key][half];
#pragma unroll
      for (int i = 0; i < 4; ++i) kdst[i] = ksrc[i];
#endif
      union { uint4 q[4]; unsigned short s[32]; } vv;
      const uint4* vsrc =
          (const uint4*)(V + ((size_t)bh * Tq + kg0 + key) * HD + half);
#pragma unroll
      for (int i = 0; i < 4; ++i) vv.q[i] = vsrc[i];
#pragma unroll
      for (int j = 0; j < 32; ++j) Vt[half + j][key] = vv.s[j];
#if HAS_ASYNC_LDS
      __builtin_amdgcn_s_wait_asynccnt(0);
#endif
    }
    __syncthreads();

    // S = Q K^T : 16 q rows x 64 keys (C layout: lane=key, vgpr=row)
    v8f s[4] = {};
#pragma unroll
    for (int kt = 0; kt < 4; ++kt)
#pragma unroll
      for (int kc = 0; kc < 2; ++kc)
        s[kt] = wmma_bf16(qa[kc], load_frag(&Ks[kt * 16 + lm][kc * 32]), s[kt]);

    // scale + causal mask
#pragma unroll
    for (int kt = 0; kt < 4; ++kt) {
      const int kg = kg0 + kt * 16 + lm;
#pragma unroll
      for (int r = 0; r < 8; ++r) {
        float sv = s[kt][r] * 0.125f;          // 1/sqrt(HD)
        if (kg > qw0 + r + 8 * mhalf) sv = -3.0e38f;
        s[kt][r] = sv;
      }
    }

    // online softmax: row max across the 16 key-lanes of each half
    float nm[8];
#pragma unroll
    for (int r = 0; r < 8; ++r) {
      float m = row_max[r];
#pragma unroll
      for (int kt = 0; kt < 4; ++kt) m = fmaxf(m, s[kt][r]);
      m = fmaxf(m, __shfl_xor(m, 1, 32));
      m = fmaxf(m, __shfl_xor(m, 2, 32));
      m = fmaxf(m, __shfl_xor(m, 4, 32));
      m = fmaxf(m, __shfl_xor(m, 8, 32));
      nm[r] = m;
      const float f = __expf(row_max[r] - m);
      row_max[r] = m;
      row_sum[r] *= f;
#pragma unroll
      for (int t = 0; t < 4; ++t) o[t][r] *= f;
    }

    // P = exp(S - max), accumulate row sums, spill P to LDS (layout change)
#pragma unroll
    for (int r = 0; r < 8; ++r) {
      float ps = 0.0f;
#pragma unroll
      for (int kt = 0; kt < 4; ++kt) {
        const float p = __expf(s[kt][r] - nm[r]);
        ps += p;
        Ps[wave][r + 8 * mhalf][kt * 16 + lm] = f2bf(p);
      }
      ps += __shfl_xor(ps, 1, 32);
      ps += __shfl_xor(ps, 2, 32);
      ps += __shfl_xor(ps, 4, 32);
      ps += __shfl_xor(ps, 8, 32);
      row_sum[r] += ps;
    }
    __syncthreads();

    // O += P @ V
#pragma unroll
    for (int pc = 0; pc < 2; ++pc) {
      const v16bf pa = load_frag(&Ps[wave][lm][pc * 32]);
#pragma unroll
      for (int ht = 0; ht < 4; ++ht)
        o[ht] = wmma_bf16(pa, load_frag(&Vt[ht * 16 + lm][pc * 32]), o[ht]);
    }
    __syncthreads();
  }

  // normalize + store ctx bf16 as [b, t, h*HD + hd]
#pragma unroll
  for (int ht = 0; ht < 4; ++ht) {
#pragma unroll
    for (int r = 0; r < 8; ++r) {
      const int t = qw0 + r + 8 * mhalf;
      const int col = h * HD + ht * 16 + lm;
      Ctx[((size_t)bb * Tq + t) * DOUT + col] =
          f2bf(o[ht][r] * (1.0f / row_sum[r]));
    }
  }
}

// ---------------------------------------------------------------------------
// Stage 3: out = ctx(bf16) @ w_o(f32->bf16) + b_o, f32 row-major output.
// ---------------------------------------------------------------------------
__global__ __launch_bounds__(256)
void out_proj_kernel(const unsigned short* __restrict__ Ctx,
                     const float* __restrict__ W,
                     const float* __restrict__ bias,
                     float* __restrict__ Out) {
  __shared__ unsigned short As[128][32];
  __shared__ unsigned short Bs[128][32];

  const int tid = threadIdx.x;
  const int lane = tid & 31, wave = tid >> 5;
  const int lm = lane & 15, mhalf = lane >> 4;
  const int wm = (wave & 3) * 32;
  const int wn = (wave >> 2) * 64;
  const int m0 = blockIdx.y * 128, n0 = blockIdx.x * 128;

  const int arow = tid >> 1, acol = (tid & 1) * 16;
  const int wrow = tid >> 3, wcol = (tid & 7) * 16;

  v8f c[2][4] = {};

  for (int kk = 0; kk < DOUT; kk += 32) {
    { // stage ctx tile (already bf16)
      const uint4* src =
          (const uint4*)(Ctx + (size_t)(m0 + arow) * DOUT + kk + acol);
      uint4* dst = (uint4*)&As[arow][acol];
      dst[0] = src[0];
      dst[1] = src[1];
    }
    { // stage W tile (32x128) transposed
      const float* src = W + (size_t)(kk + wrow) * DOUT + n0 + wcol;
      float vals[16];
#pragma unroll
      for (int i = 0; i < 4; ++i) {
        float4 f = ((const float4*)src)[i];
        vals[i * 4 + 0] = f.x; vals[i * 4 + 1] = f.y;
        vals[i * 4 + 2] = f.z; vals[i * 4 + 3] = f.w;
      }
#pragma unroll
      for (int j = 0; j < 16; ++j) Bs[wcol + j][wrow] = f2bf(vals[j]);
    }
    __syncthreads();

    v16bf a0 = load_frag(&As[wm + lm][0]);
    v16bf a1 = load_frag(&As[wm + 16 + lm][0]);
#pragma unroll
    for (int j = 0; j < 4; ++j) {
      const v16bf b = load_frag(&Bs[wn + j * 16 + lm][0]);
      c[0][j] = wmma_bf16(a0, b, c[0][j]);
      c[1][j] = wmma_bf16(a1, b, c[1][j]);
    }
    __syncthreads();
  }

#pragma unroll
  for (int j = 0; j < 4; ++j) {
    const int gn = n0 + wn + j * 16 + lm;
    const float bj = bias[gn];
#pragma unroll
    for (int i = 0; i < 2; ++i) {
#pragma unroll
      for (int r = 0; r < 8; ++r) {
        const int gm = m0 + wm + i * 16 + r + 8 * mhalf;
        Out[(size_t)gm * DOUT + gn] = c[i][j][r] + bj;
      }
    }
  }
}

// ---------------------------------------------------------------------------
extern "C" void kernel_launch(void* const* d_in, const int* in_sizes, int n_in,
                              void* d_out, int out_size, void* d_ws,
                              size_t ws_size, hipStream_t stream) {
  const float* x   = (const float*)d_in[0];
  const float* w_q = (const float*)d_in[1];
  const float* b_q = (const float*)d_in[2];
  const float* w_k = (const float*)d_in[3];
  const float* b_k = (const float*)d_in[4];
  const float* w_v = (const float*)d_in[5];
  const float* b_v = (const float*)d_in[6];
  const float* w_o = (const float*)d_in[7];
  const float* b_o = (const float*)d_in[8];
  float* out = (float*)d_out;

  const size_t NTOK = 4ull * Tq;                      // 8192 tokens
  unsigned short* qws = (unsigned short*)d_ws;        // [B,H,T,HD] bf16
  unsigned short* kws = qws + NTOK * DOUT;
  unsigned short* vws = kws + NTOK * DOUT;
  unsigned short* cws = vws + NTOK * DOUT;            // [B,T,D] bf16

  dim3 gproj(DOUT / 128, NTOK / 128), bproj(256);
  proj_qkv_kernel<<<gproj, bproj, 0, stream>>>(x, w_q, b_q, qws);
  proj_qkv_kernel<<<gproj, bproj, 0, stream>>>(x, w_k, b_k, kws);
  proj_qkv_kernel<<<gproj, bproj, 0, stream>>>(x, w_v, b_v, vws);

  dim3 gattn(Tq / 64, 4 * Hh), battn(128);
  flash_attn_kernel<<<gattn, battn, 0, stream>>>(qws, kws, vws, cws);

  out_proj_kernel<<<gproj, bproj, 0, stream>>>(cws, w_o, b_o, out);
}
